// MSARowAttentionWithBias_58548994179788
// MI455X (gfx1250) — matrix operand order repaired and checked by reading, not compiled, not run on hardware
//
#include <hip/hip_runtime.h>
#include <hip/hip_bf16.h>

// ---------------------------------------------------------------------------
// MSARowAttentionWithBias for MI455X (gfx1250) — WMMA f16 pipeline, v3.
// A tiles staged via GLOBAL_LOAD_ASYNC_TO_LDS (ASYNCcnt); 2 WMMA tiles/wave.
// ---------------------------------------------------------------------------

typedef _Float16 half_t;
typedef __attribute__((ext_vector_type(16))) _Float16 v16h;
typedef __attribute__((ext_vector_type(8)))  _Float16 v8h;
typedef __attribute__((ext_vector_type(8)))  float    v8f;

#define NSEQ  256   // N sequences
#define LRES  384   // L residues
#define DMSA  256   // d_msa
#define DPAIR 128   // d_pair
#define NH    8     // heads
#define DHD   32    // head dim
#define HD    256   // NH*DHD

__device__ __forceinline__ v8f wmma_f16(v16h a, v16h b, v8f c) {
    return __builtin_amdgcn_wmma_f32_16x16x32_f16(
        /*neg_a=*/false, a, /*neg_b=*/false, b,
        /*c_mod=*/(short)0, c, /*reuse_a=*/false, /*reuse_b=*/false);
}

// Async copy of 16 bytes global -> LDS (no VGPR bounce). LDS address is the
// low 32 bits of the generic pointer (flat LDS aperture, ISA 10.2).
__device__ __forceinline__ void async_copy_b128(const half_t* gsrc,
                                                half_t* lds_dst) {
    const unsigned lds = (unsigned)(size_t)lds_dst;
    asm volatile("global_load_async_to_lds_b128 %0, %1, off"
                 :: "v"(lds), "v"(gsrc) : "memory");
}
__device__ __forceinline__ void wait_async() {
    asm volatile("s_wait_asynccnt 0x0" ::: "memory");
}

// A fragment: 16 (M) x 32 (K), row-major, leading dim lda (multiple of 8).
__device__ __forceinline__ v16h load_a_frag(const half_t* __restrict__ A,
                                            int lda, int lane) {
    const half_t* p = A + (size_t)(lane & 15) * lda + ((lane >> 4) * 8);
    const v8h lo = *(const v8h*)p;
    const v8h hi = *(const v8h*)(p + 16);
    return __builtin_shufflevector(lo, hi, 0, 1, 2, 3, 4, 5, 6, 7,
                                           8, 9, 10, 11, 12, 13, 14, 15);
}

// B fragment from a column-major view (element [k][col] = Bt[col*ldcol + k]).
__device__ __forceinline__ v16h load_bt_frag(const half_t* __restrict__ Bt,
                                             int ldcol, int lane) {
    const half_t* p = Bt + (size_t)(lane & 15) * ldcol + (lane >> 4) * 8;
    const v8h lo = *(const v8h*)p;
    const v8h hi = *(const v8h*)(p + 16);
    return __builtin_shufflevector(lo, hi, 0, 1, 2, 3, 4, 5, 6, 7,
                                           8, 9, 10, 11, 12, 13, 14, 15);
}

__device__ __forceinline__ float wave_sum(float v) {
#pragma unroll
    for (int o = 16; o > 0; o >>= 1) v += __shfl_xor(v, o, 32);
    return v;
}
__device__ __forceinline__ float wave_max(float v) {
#pragma unroll
    for (int o = 16; o > 0; o >>= 1) v = fmaxf(v, __shfl_xor(v, o, 32));
    return v;
}

// ---------------------------------------------------------------------------
// 0. f32 -> f16 weight conversion with transpose: dst[n,k] = src[k,n]
// ---------------------------------------------------------------------------
__global__ void cvt_t_f16_kernel(const float* __restrict__ src,
                                 half_t* __restrict__ dst, int K, int N) {
    int i = blockIdx.x * 256 + threadIdx.x;
    if (i < K * N) {
        int k = i / N, n = i - k * N;
        dst[(size_t)n * K + k] = (half_t)src[i];
    }
}

// ---------------------------------------------------------------------------
// 1. LayerNorm over msa rows (D=256), 256 threads/row, output f16
// ---------------------------------------------------------------------------
__global__ void ln_msa_kernel(const float* __restrict__ msa,
                              const float* __restrict__ g,
                              const float* __restrict__ b,
                              half_t* __restrict__ mh) {
    const int row = blockIdx.x;          // n*L + i
    const int t   = threadIdx.x;         // 0..255
    const int wv  = t >> 5, ln = t & 31;
    __shared__ float sred[8];
    float x = msa[(size_t)row * DMSA + t];
    float s = wave_sum(x);
    if (ln == 0) sred[wv] = s;
    __syncthreads();
    float tot = 0.f;
#pragma unroll
    for (int i = 0; i < 8; ++i) tot += sred[i];
    const float mu = tot * (1.0f / DMSA);
    const float d  = x - mu;
    __syncthreads();
    float s2 = wave_sum(d * d);
    if (ln == 0) sred[wv] = s2;
    __syncthreads();
    float tot2 = 0.f;
#pragma unroll
    for (int i = 0; i < 8; ++i) tot2 += sred[i];
    const float rstd = rsqrtf(tot2 * (1.0f / DMSA) + 1e-5f);
    mh[(size_t)row * DMSA + t] = (half_t)(d * rstd * g[t] + b[t]);
}

// ---------------------------------------------------------------------------
// 2. Generic WMMA GEMM: C(MxN) = A(MxK,row-major) @ B, Bt = B^T (NxK).
//    Block: 8 waves, each wave computes a 16x32 strip (2 accumulators sharing
//    one A fragment) -> block covers 16 rows x 256 cols. A tile staged via
//    async global->LDS.
//    mode 0: outh = (acc + bias)*scale                (f16, row-major)
//    mode 1: outh = sigmoid(acc + bias)               (f16, row-major)
//    mode 2: outf = (acc + bias)*scale                (f32, row-major)
//    mode 3: outh[((n*NH+h)*DHD+d)*LRES + j] = acc    (f16, per-head V^T)
// ---------------------------------------------------------------------------
__global__ void gemm_wmma_kernel(const half_t* __restrict__ A,
                                 const half_t* __restrict__ Bt,
                                 const float* __restrict__ bias,
                                 half_t* __restrict__ outh,
                                 float* __restrict__ outf,
                                 int M, int K, int Nn, int mode, float scale) {
    __shared__ __align__(16) half_t Atile[16 * 32];
    const int t    = threadIdx.x;
    const int lane = t & 31;
    const int wave = t >> 5;
    const int row0 = blockIdx.x * 16;
    const int col0 = blockIdx.y * 256 + wave * 32;
    const bool active = (col0 < Nn);     // wave-uniform

    v8f acc[2] = {{}, {}};
    for (int k0 = 0; k0 < K; k0 += 32) {
        if (t < 64) {                    // async-stage A 16x32 tile (1KB)
            const int r = t >> 2, cc = (t & 3) * 8;
            async_copy_b128(A + (size_t)(row0 + r) * K + k0 + cc,
                            &Atile[r * 32 + cc]);
        }
        wait_async();
        __syncthreads();
        if (active) {
            v16h a = load_a_frag(Atile, 32, lane);
#pragma unroll
            for (int tt = 0; tt < 2; ++tt) {
                const half_t* bp = Bt + (size_t)(col0 + tt * 16) * K + k0;
                v16h b = load_bt_frag(bp, K, lane);
                if (k0 + 32 < K) __builtin_prefetch(bp + 32, 0, 3);
                acc[tt] = wmma_f16(a, b, acc[tt]);
            }
        }
        __syncthreads();
    }
    if (!active) return;

    const int rb = row0 + ((lane >> 4) * 8);
#pragma unroll
    for (int tt = 0; tt < 2; ++tt) {
        const int c  = col0 + tt * 16 + (lane & 15);
        const float bv = bias ? bias[c] : 0.0f;
#pragma unroll
        for (int r = 0; r < 8; ++r) {
            const int   row = rb + r;
            const float x   = acc[tt][r] + bv;
            if (mode == 0)      outh[(size_t)row * Nn + c] = (half_t)(x * scale);
            else if (mode == 1) outh[(size_t)row * Nn + c] =
                                    (half_t)(1.0f / (1.0f + __expf(-x)));
            else if (mode == 2) outf[(size_t)row * Nn + c] = x * scale;
            else {               // V^T per-head store
                const int n = row / LRES, j = row - n * LRES;
                const int h = c >> 5,     d = c & 31;
                outh[(((size_t)n * NH + h) * DHD + d) * LRES + j] =
                    (half_t)(x * scale);
            }
        }
    }
}

// ---------------------------------------------------------------------------
// 3. SequenceWeight logits: swl[n,i,h] = dot_d(qsw[i,h,:], ksw[n,i,h,:])
// ---------------------------------------------------------------------------
__global__ void swlogits_kernel(const float* __restrict__ qsw,
                                const half_t* __restrict__ ksw,
                                float* __restrict__ swl) {
    const size_t gid   = (size_t)blockIdx.x * 256 + threadIdx.x; // (n*L+i)*H+h
    const size_t total = (size_t)NSEQ * LRES * NH;
    if (gid >= total) return;
    const int h = (int)(gid % NH);
    const int i = (int)((gid / NH) % LRES);
    const float*  qp = qsw + (size_t)i * HD + h * DHD;
    const half_t* kp = ksw + gid * DHD;
    float acc = 0.f;
#pragma unroll
    for (int d = 0; d < DHD; ++d) acc += qp[d] * (float)kp[d];
    swl[gid] = acc;
}

// ---------------------------------------------------------------------------
// 4. Softmax over n (256 values) per (i,h)
// ---------------------------------------------------------------------------
__global__ void seqw_softmax_kernel(const float* __restrict__ swl,
                                    float* __restrict__ sw) {
    const int bid = blockIdx.x;          // i*NH + h
    const int i = bid / NH, h = bid % NH;
    const int n = threadIdx.x;           // 0..255
    __shared__ float sred[8];
    const size_t idx = ((size_t)n * LRES + i) * NH + h;
    const float x = swl[idx];
    float m = wave_max(x);
    if ((n & 31) == 0) sred[n >> 5] = m;
    __syncthreads();
    float bm = sred[0];
#pragma unroll
    for (int k = 1; k < 8; ++k) bm = fmaxf(bm, sred[k]);
    const float e = __expf(x - bm);
    __syncthreads();
    float s = wave_sum(e);
    if ((n & 31) == 0) sred[n >> 5] = s;
    __syncthreads();
    float bs = 0.f;
#pragma unroll
    for (int k = 0; k < 8; ++k) bs += sred[k];
    sw[idx] = e / bs;
}

// ---------------------------------------------------------------------------
// 5. q *= seq_weight (broadcast over d)
// ---------------------------------------------------------------------------
__global__ void apply_seqw_kernel(half_t* __restrict__ qh,
                                  const float* __restrict__ sw) {
    const size_t gid   = (size_t)blockIdx.x * 256 + threadIdx.x;
    const size_t total = (size_t)NSEQ * LRES * HD;
    if (gid >= total) return;
    qh[gid] = (half_t)((float)qh[gid] * sw[gid >> 5]);
}

// ---------------------------------------------------------------------------
// 6. logits[h,i,j] = sum_{n,d} q[n,i,h,d]*k[n,j,h,d]. One WMMA per n
//    (DH=32 contiguous contraction). A tile async-staged in LDS.
// ---------------------------------------------------------------------------
__global__ void logits_wmma_kernel(const half_t* __restrict__ qh,
                                   const half_t* __restrict__ kh,
                                   float* __restrict__ logits) {
    __shared__ __align__(16) half_t Atile[16 * 32];
    const int t    = threadIdx.x;
    const int lane = t & 31;
    const int wave = t >> 5;
    const int h    = blockIdx.z;
    const int row0 = blockIdx.x * 16;                 // i tile (24)
    const int col0 = (blockIdx.y * 8 + wave) * 16;    // j tile (24)

    v8f acc = {};
    for (int n = 0; n < NSEQ; ++n) {
        if (t < 64) {
            const int r = t >> 2, cc = (t & 3) * 8;
            async_copy_b128(
                qh + ((size_t)(n * LRES + row0 + r) * NH + h) * DHD + cc,
                &Atile[r * 32 + cc]);
        }
        wait_async();
        __syncthreads();
        v16h a = load_a_frag(Atile, 32, lane);
        const half_t* bp = kh + ((size_t)(n * LRES + col0) * NH + h) * DHD;
        v16h b = load_bt_frag(bp, NH * DHD, lane);
        if (n + 1 < NSEQ)
            __builtin_prefetch(
                kh + ((size_t)((n + 1) * LRES + col0) * NH + h) * DHD, 0, 3);
        acc = wmma_f16(a, b, acc);
        __syncthreads();
    }
    const int c  = col0 + (lane & 15);
    const int rb = row0 + ((lane >> 4) * 8);
#pragma unroll
    for (int r = 0; r < 8; ++r)
        logits[((size_t)h * LRES + rb + r) * LRES + c] = acc[r];
}

// ---------------------------------------------------------------------------
// 7. Pair bias: layernorm(pair[i,j,:]) @ Wb -> biasb[h,i,j]
// ---------------------------------------------------------------------------
__global__ void pair_bias_kernel(const float* __restrict__ pair,
                                 const float* __restrict__ g,
                                 const float* __restrict__ b,
                                 const float* __restrict__ Wb,
                                 float* __restrict__ biasb) {
    const int pr = blockIdx.x;           // i*L + j
    const int t  = threadIdx.x;          // 0..127
    const int i = pr / LRES, j = pr - (pr / LRES) * LRES;
    __shared__ float sred[4];
    __shared__ float sh[DPAIR];
    const float x = pair[(size_t)pr * DPAIR + t];
    float s = wave_sum(x);
    if ((t & 31) == 0) sred[t >> 5] = s;
    __syncthreads();
    const float mu = (sred[0] + sred[1] + sred[2] + sred[3]) * (1.0f / DPAIR);
    const float d  = x - mu;
    __syncthreads();
    float s2 = wave_sum(d * d);
    if ((t & 31) == 0) sred[t >> 5] = s2;
    __syncthreads();
    const float var = (sred[0] + sred[1] + sred[2] + sred[3]) * (1.0f / DPAIR);
    sh[t] = d * rsqrtf(var + 1e-5f) * g[t] + b[t];
    __syncthreads();
    if (t < NH) {
        float acc = 0.f;
        for (int c = 0; c < DPAIR; ++c) acc += sh[c] * Wb[c * NH + t];
        biasb[((size_t)t * LRES + i) * LRES + j] = acc;
    }
}

// ---------------------------------------------------------------------------
// 8. Softmax over j (384) of logits+bias -> attn f16 [h,i,j]
// ---------------------------------------------------------------------------
__global__ void attn_softmax_kernel(const float* __restrict__ logits,
                                    const float* __restrict__ biasb,
                                    half_t* __restrict__ attnh) {
    const int bid = blockIdx.x;          // h*L + i
    const int j   = threadIdx.x;         // 0..383 (12 waves)
    __shared__ float sred[12];
    const size_t base = (size_t)bid * LRES;
    const float x = logits[base + j] + biasb[base + j];
    float m = wave_max(x);
    if ((j & 31) == 0) sred[j >> 5] = m;
    __syncthreads();
    float bm = sred[0];
#pragma unroll
    for (int k = 1; k < 12; ++k) bm = fmaxf(bm, sred[k]);
    const float e = __expf(x - bm);
    __syncthreads();
    float s = wave_sum(e);
    if ((j & 31) == 0) sred[j >> 5] = s;
    __syncthreads();
    float bs = 0.f;
#pragma unroll
    for (int k = 0; k < 12; ++k) bs += sred[k];
    attnh[base + j] = (half_t)(e / bs);
}

// ---------------------------------------------------------------------------
// 9. out[n,i,h,:] = attn_h(LxL) @ V_{n,h}(LxDH), fused *gate -> f16.
//    Each wave computes a full 16x32 strip (both d-tiles, shared A fragment).
//    Block = 8 waves = 128 rows of attn for one (n,h).
// ---------------------------------------------------------------------------
__global__ void av_wmma_kernel(const half_t* __restrict__ attnh,
                               const half_t* __restrict__ vt,
                               const half_t* __restrict__ gh,
                               half_t* __restrict__ oh) {
    const int lane = threadIdx.x & 31;
    const int wave = threadIdx.x >> 5;
    const int n = blockIdx.y, h = blockIdx.z;
    const int row0 = (blockIdx.x * 8 + wave) * 16;         // i tile (24)

    v8f acc[2] = {{}, {}};
    for (int k0 = 0; k0 < LRES; k0 += 32) {
        v16h a = load_a_frag(attnh + ((size_t)h * LRES + row0) * LRES + k0,
                             LRES, lane);
#pragma unroll
        for (int tt = 0; tt < 2; ++tt) {
            const half_t* bp =
                vt + (((size_t)n * NH + h) * DHD + tt * 16) * LRES + k0;
            v16h b = load_bt_frag(bp, LRES, lane);
            if (k0 + 32 < LRES) __builtin_prefetch(bp + 32, 0, 3);
            acc[tt] = wmma_f16(a, b, acc[tt]);
        }
    }
    const int rb = row0 + ((lane >> 4) * 8);
#pragma unroll
    for (int tt = 0; tt < 2; ++tt) {
        const int c = tt * 16 + (lane & 15);
#pragma unroll
        for (int r = 0; r < 8; ++r) {
            const size_t idx = ((size_t)(n * LRES + rb + r) * NH + h) * DHD + c;
            oh[idx] = (half_t)(acc[tt][r] * (float)gh[idx]);
        }
    }
}

// ---------------------------------------------------------------------------
// Host launch
// ---------------------------------------------------------------------------
extern "C" void kernel_launch(void* const* d_in, const int* in_sizes, int n_in,
                              void* d_out, int out_size, void* d_ws, size_t ws_size,
                              hipStream_t stream) {
    const float* msa      = (const float*)d_in[0];
    const float* pair     = (const float*)d_in[1];
    const float* ln_msa_g = (const float*)d_in[2];
    const float* ln_msa_b = (const float*)d_in[3];
    const float* ln_pr_g  = (const float*)d_in[4];
    const float* ln_pr_b  = (const float*)d_in[5];
    const float* Wq_sw    = (const float*)d_in[6];
    const float* bq_sw    = (const float*)d_in[7];
    const float* Wk_sw    = (const float*)d_in[8];
    const float* bk_sw    = (const float*)d_in[9];
    const float* Wq       = (const float*)d_in[10];
    const float* Wk       = (const float*)d_in[11];
    const float* Wv       = (const float*)d_in[12];
    const float* Wb       = (const float*)d_in[13];
    const float* Wg       = (const float*)d_in[14];
    const float* bg       = (const float*)d_in[15];
    const float* Wo       = (const float*)d_in[16];
    const float* bo       = (const float*)d_in[17];
    float* out = (float*)d_out;

    const size_t ML    = (size_t)NSEQ * LRES;         // 98304 rows
    const size_t WSZ   = (size_t)DMSA * HD;           // 65536 weight elems
    const float  scale = 0.17677669529663687f;        // 1/sqrt(32)

    // ---- workspace carve (256B aligned) ----
    size_t off = 0;
    auto carve = [&](size_t bytes) -> char* {
        char* p = (char*)d_ws + off;
        off += (bytes + 255) & ~(size_t)255;
        return p;
    };
    half_t* Whqsw = (half_t*)carve(WSZ * 2);   // all weights stored TRANSPOSED
    half_t* Whksw = (half_t*)carve(WSZ * 2);
    half_t* Whq   = (half_t*)carve(WSZ * 2);
    half_t* Whk   = (half_t*)carve(WSZ * 2);
    half_t* Whv   = (half_t*)carve(WSZ * 2);
    half_t* Whg   = (half_t*)carve(WSZ * 2);
    half_t* Who   = (half_t*)carve(WSZ * 2);
    half_t* mh    = (half_t*)carve(ML * DMSA * 2);
    half_t* kswh  = (half_t*)carve(ML * HD * 2);
    half_t* qh    = (half_t*)carve(ML * HD * 2);
    half_t* kh    = (half_t*)carve(ML * HD * 2);
    half_t* vt    = (half_t*)carve(ML * HD * 2);      // [n,h,d,j] transposed V
    half_t* gh    = (half_t*)carve(ML * HD * 2);
    float*  qswf  = (float*)carve((size_t)LRES * HD * 4);
    float*  swl   = (float*)carve(ML * NH * 4);
    float*  swf   = (float*)carve(ML * NH * 4);
    float*  lgf   = (float*)carve((size_t)NH * LRES * LRES * 4);
    float*  bbf   = (float*)carve((size_t)NH * LRES * LRES * 4);
    half_t* attnh = (half_t*)carve((size_t)NH * LRES * LRES * 2);
    half_t* oh    = kswh;   // ksw is dead after seq-weight logits; reuse.
    (void)ws_size; (void)in_sizes; (void)n_in; (void)out_size;

    // ---- 0. weight conversion f32 -> f16, transposed ----
    const int cb = (int)((WSZ + 255) / 256);
    cvt_t_f16_kernel<<<cb, 256, 0, stream>>>(Wq_sw, Whqsw, DMSA, HD);
    cvt_t_f16_kernel<<<cb, 256, 0, stream>>>(Wk_sw, Whksw, DMSA, HD);
    cvt_t_f16_kernel<<<cb, 256, 0, stream>>>(Wq,    Whq,   DMSA, HD);
    cvt_t_f16_kernel<<<cb, 256, 0, stream>>>(Wk,    Whk,   DMSA, HD);
    cvt_t_f16_kernel<<<cb, 256, 0, stream>>>(Wv,    Whv,   DMSA, HD);
    cvt_t_f16_kernel<<<cb, 256, 0, stream>>>(Wg,    Whg,   DMSA, HD);
    cvt_t_f16_kernel<<<cb, 256, 0, stream>>>(Wo,    Who,   HD,   DMSA);

    // ---- 1. layernorm msa -> mh (f16) ----
    ln_msa_kernel<<<(int)ML, 256, 0, stream>>>(msa, ln_msa_g, ln_msa_b, mh);

    // ---- 2. projections (WMMA GEMMs, Bt = transposed weights) ----
    dim3 gBig((unsigned)(ML / 16), 1);      // block covers 16 rows x 256 cols
    dim3 gQsw((unsigned)(LRES / 16), 1);
    // k_sw = m @ Wk_sw + bk_sw (f16)
    gemm_wmma_kernel<<<gBig, 256, 0, stream>>>(mh, Whksw, bk_sw, kswh, nullptr,
                                               (int)ML, DMSA, HD, 0, 1.0f);
    // q_sw = (m[0] @ Wq_sw + bq_sw)*scale (f32; rows 0..L-1 of mh are n=0)
    gemm_wmma_kernel<<<gQsw, 256, 0, stream>>>(mh, Whqsw, bq_sw, nullptr, qswf,
                                               LRES, DMSA, HD, 2, scale);
    // seq-weight logits + softmax over n
    swlogits_kernel<<<(int)((ML * NH + 255) / 256), 256, 0, stream>>>(qswf, kswh, swl);
    seqw_softmax_kernel<<<LRES * NH, 256, 0, stream>>>(swl, swf);
    // q = m @ Wq (f16), then q *= seq_weight
    gemm_wmma_kernel<<<gBig, 256, 0, stream>>>(mh, Whq, nullptr, qh, nullptr,
                                               (int)ML, DMSA, HD, 0, 1.0f);
    apply_seqw_kernel<<<(int)((ML * HD + 255) / 256), 256, 0, stream>>>(qh, swf);
    // k = (m @ Wk)*scale (f16), v = m @ Wv (f16, per-head transposed),
    // gate = sigmoid(m @ Wg + bg) (f16)
    gemm_wmma_kernel<<<gBig, 256, 0, stream>>>(mh, Whk, nullptr, kh, nullptr,
                                               (int)ML, DMSA, HD, 0, scale);
    gemm_wmma_kernel<<<gBig, 256, 0, stream>>>(mh, Whv, nullptr, vt, nullptr,
                                               (int)ML, DMSA, HD, 3, 1.0f);
    gemm_wmma_kernel<<<gBig, 256, 0, stream>>>(mh, Whg, bg, gh, nullptr,
                                               (int)ML, DMSA, HD, 1, 1.0f);

    // ---- 3. logits[h] = Q_h K_h^T  (K = N*DH = 8192) ----
    dim3 gLog(LRES / 16, 3, NH);            // 24 x 24 tiles x 8 heads
    logits_wmma_kernel<<<gLog, 256, 0, stream>>>(qh, kh, lgf);

    // ---- 4. pair bias + softmax over j ----
    pair_bias_kernel<<<LRES * LRES, DPAIR, 0, stream>>>(pair, ln_pr_g, ln_pr_b,
                                                        Wb, bbf);
    attn_softmax_kernel<<<NH * LRES, LRES, 0, stream>>>(lgf, bbf, attnh);

    // ---- 5. out = (attn @ v) * gate (f16) ----
    dim3 gAV(LRES / 128, NSEQ, NH);         // 3 x 256 x 8 blocks of 8 waves
    av_wmma_kernel<<<gAV, 256, 0, stream>>>(attnh, vt, gh, oh);

    // ---- 6. final projection: out @ Wo + bo -> f32 d_out ----
    gemm_wmma_kernel<<<gBig, 256, 0, stream>>>(oh, Who, bo, nullptr, out,
                                               (int)ML, HD, DMSA, 2, 1.0f);
}
